// MultiHeadAttention_74019466380032
// MI455X (gfx1250) — matrix operand order repaired and checked by reading, not compiled, not run on hardware
//
#include <hip/hip_runtime.h>

// Problem constants (from reference)
#define CC    6
#define HH    8
#define UU    100
#define DD    512
#define TT    512
#define BB    16
#define UPAD  128          // U padded to multiple of 32 for WMMA K / N tiling
#define OUTF  800          // H*U
#define OPAD  112          // U padded to multiple of 16 for output N tiling
#define MROWS 8192         // B*T

typedef __attribute__((ext_vector_type(16))) __bf16 v16bf;
typedef __attribute__((ext_vector_type(8)))  float  v8f;

#if __has_builtin(__builtin_amdgcn_global_load_async_to_lds_b128) && \
    __has_builtin(__builtin_amdgcn_s_wait_asynccnt)
#define USE_ASYNC_LDS 1
#else
#define USE_ASYNC_LDS 0
#endif

// ---------------- 16B-vectorized fragment load helpers ----------------

__device__ __forceinline__ v16bf load16(const __bf16* p0, const __bf16* p1) {
  union { uint4 q[2]; v16bf v; } u;
  u.q[0] = *(const uint4*)p0;
  u.q[1] = *(const uint4*)p1;
  return u.v;
}

__device__ __forceinline__ v8f wmma_bf16(v16bf a, v16bf b, v8f c) {
  return __builtin_amdgcn_wmma_f32_16x16x32_bf16(false, a, false, b, (short)0, c,
                                                 false, false);
}

// A matrix 16x32 (MxK), row-major source with row stride ldm (16B-aligned rows).
// lanes 0-15: M=lane, K {half*8..+7, half*8+16..+23}
__device__ __forceinline__ v16bf load_a_frag(const __bf16* base, int ldm) {
  int lane = threadIdx.x & 31;
  int m    = lane & 15;
  int half = lane >> 4;
  const __bf16* p = base + (size_t)m * ldm + half * 8;
  return load16(p, p + 16);
}

// B matrix 32x16 (KxN) where memory is K-major per column: element (k,n) at
// base[n*ldn + k]. lanes 0-15: N=lane, K {half*8..+7, half*8+16..+23}.
// Per lane: two 16B runs.
__device__ __forceinline__ v16bf load_bk_frag(const __bf16* base, int ldn) {
  int lane = threadIdx.x & 31;
  int n    = lane & 15;
  int kh   = (lane >> 4) * 8;
  const __bf16* p = base + (size_t)n * ldn + kh;
  return load16(p, p + 16);
}

// D/C 16x16 f32: lanes 0-15: N=lane, M=vgpr ; lanes 16-31: N=lane-16, M=8+vgpr
__device__ __forceinline__ void store_d_f32(float* base, int ld, v8f d) {
  int lane = threadIdx.x & 31;
  int n    = lane & 15;
  int mh   = (lane >> 4) * 8;
#pragma unroll
  for (int i = 0; i < 8; ++i) base[(size_t)(mh + i) * ld + n] = d[i];
}

__device__ __forceinline__ void store_d_bf16(__bf16* base, int ld, v8f d) {
  int lane = threadIdx.x & 31;
  int n    = lane & 15;
  int mh   = (lane >> 4) * 8;
#pragma unroll
  for (int i = 0; i < 8; ++i) base[(size_t)(mh + i) * ld + n] = (__bf16)d[i];
}

// Stage `n16` 16-byte chunks from global to LDS (contiguous), all threads.
__device__ __forceinline__ void stage_tile(const __bf16* gsrc, __bf16* lds, int n16,
                                           int nthreads) {
#if USE_ASYNC_LDS
  typedef int vint4 __attribute__((vector_size(16)));   // clang native vector, not HIP int4
  typedef __attribute__((address_space(1))) vint4 GV4;
  typedef __attribute__((address_space(3))) vint4 LV4;
  GV4* g = (GV4*)(unsigned long long)gsrc;
  // generic LDS address: low 32 bits are the AS(3) offset
  LV4* l = (LV4*)(unsigned int)(unsigned long long)lds;
  for (int i = threadIdx.x; i < n16; i += nthreads)
    __builtin_amdgcn_global_load_async_to_lds_b128(g + i, l + i, 0, 0);
  __builtin_amdgcn_s_wait_asynccnt(0);
#else
  const uint4* src = (const uint4*)gsrc;
  uint4*       dst = (uint4*)lds;
  for (int i = threadIdx.x; i < n16; i += nthreads) dst[i] = src[i];
#endif
}

// ---------------- prep kernels ----------------

__global__ void cvt_bf16_kernel(const float* __restrict__ in, __bf16* __restrict__ out, int n) {
  int i = blockIdx.x * 256 + threadIdx.x;
  if (i < n) out[i] = (__bf16)in[i];
}

// Transposed fused weight: Wt[g][u][d] = sum_w WQ[g][d][w] * Wdot[g][w][u] (0 for u>=100)
__global__ void prep_wqd_kernel(const float* __restrict__ WQ, const float* __restrict__ Wd,
                                __bf16* __restrict__ out) {
  int idx = blockIdx.x * 256 + threadIdx.x;
  if (idx >= CC * HH * UPAD * DD) return;
  int d = idx & (DD - 1);
  int u = (idx >> 9) & (UPAD - 1);
  int g = idx >> 16;
  float acc = 0.f;
  if (u < UU) {
    const float* wq = WQ + ((size_t)g * DD + d) * UU;
    const float* wd = Wd + (size_t)g * UU * UU + u;
    for (int w2 = 0; w2 < UU; ++w2) acc += wq[w2] * wd[(size_t)w2 * UU];
  }
  out[idx] = (__bf16)acc;
}

// Transposed pad: Wt[g][u][d] = W[g][d][u]  (0 for u>=100)
__global__ void prep_wpad_kernel(const float* __restrict__ W, __bf16* __restrict__ out) {
  int idx = blockIdx.x * 256 + threadIdx.x;
  if (idx >= CC * HH * UPAD * DD) return;
  int d = idx & (DD - 1);
  int u = (idx >> 9) & (UPAD - 1);
  int g = idx >> 16;
  out[idx] = (__bf16)((u < UU) ? W[((size_t)g * DD + d) * UU + u] : 0.f);
}

// Transposed WO: WOt[c][u:OPAD][o:OUTF] = WO[c][o][u] (0 for u>=100)
__global__ void prep_wo_kernel(const float* __restrict__ WO, __bf16* __restrict__ out) {
  int idx = blockIdx.x * 256 + threadIdx.x;
  if (idx >= CC * OPAD * OUTF) return;
  int o = idx % OUTF;
  int r = idx / OUTF;
  int u = r % OPAD;
  int c = r / OPAD;
  out[idx] = (__bf16)((u < UU) ? WO[((size_t)c * OUTF + o) * UU + u] : 0.f);
}

// ---------------- Q/K projection: out[g] = x(8192x512) @ W[g]^T  ----------------
// W stored transposed [g][u][d]; grid: (mtile 512, g 48, which 2); block 256 (8 waves)

__global__ __launch_bounds__(256) void qk_kernel(
    const __bf16* __restrict__ xb,
    const __bf16* __restrict__ wqd, const __bf16* __restrict__ wk,
    __bf16* __restrict__ qw, __bf16* __restrict__ kb) {
  __shared__ alignas(16) __bf16 sx[16 * DD];  // 16 KB x-tile
  int mt = blockIdx.x, g = blockIdx.y, which = blockIdx.z;
  const __bf16* W   = (which == 0) ? wqd : wk;
  __bf16*       out = (which == 0) ? qw : kb;

  stage_tile(xb + (size_t)mt * 16 * DD, sx, (16 * DD * 2) / 16, 256);
  __syncthreads();

  int w = threadIdx.x >> 5;
  const __bf16* Wg = W + ((size_t)g * UPAD + w * 16) * DD;  // rows of W^T (u-major)
  v8f acc = {};
#pragma unroll
  for (int kc = 0; kc < DD / 32; ++kc) {
    v16bf a = load_a_frag(sx + kc * 32, DD);
    v16bf b = load_bk_frag(Wg + kc * 32, DD);
    acc = wmma_bf16(a, b, acc);
  }
  __bf16* outp = out + (size_t)g * MROWS * UPAD + (size_t)mt * 16 * UPAD + w * 16;
  store_d_bf16(outp, UPAD, acc);
}

// ---------------- V projection, transposed output: vT[g][u][M] ----------------
// vT = W^T(128x512) @ x^T(512x8192). A = Wt rows-of-u (contiguous),
// B = x^T, contiguous along d in row-major x (staged in LDS).
// grid: (ttile 512, g 48); block 256 (8 waves, wave w owns u-tile w)

__global__ __launch_bounds__(256) void vproj_kernel(
    const __bf16* __restrict__ xb, const __bf16* __restrict__ wv,
    __bf16* __restrict__ vt) {
  __shared__ alignas(16) __bf16 sx[16 * DD];  // 16 x-rows (t), used as B
  int tt = blockIdx.x, g = blockIdx.y;

  stage_tile(xb + (size_t)tt * 16 * DD, sx, (16 * DD * 2) / 16, 256);
  __syncthreads();

  int w = threadIdx.x >> 5;
  const __bf16* Wg = wv + ((size_t)g * UPAD + w * 16) * DD;
  v8f acc = {};
#pragma unroll
  for (int kc = 0; kc < DD / 32; ++kc) {
    v16bf a = load_a_frag(Wg + kc * 32, DD);     // A: 16 u-rows x 32 d
    v16bf b = load_bk_frag(sx + kc * 32, DD);    // B: x^T, contiguous along d
    acc = wmma_bf16(a, b, acc);
  }
  // D rows = u, cols = t  ->  vT[(g*UPAD + u)*MROWS + t]
  __bf16* outp = vt + ((size_t)g * UPAD + w * 16) * MROWS + (size_t)tt * 16;
  store_d_bf16(outp, MROWS, acc);
}

// ---------------- fused attention per (c,h,b, 16-row tile) ----------------
// grid: (mt 32, b 16, g 48)   block: 256 (8 waves)

__global__ __launch_bounds__(256) void attn_kernel(
    const __bf16* __restrict__ qw, const __bf16* __restrict__ kb,
    const __bf16* __restrict__ vt, __bf16* __restrict__ ob) {
  __shared__ alignas(16) float  slog[16 * TT];   // 32 KB f32 logits
  __shared__ alignas(16) __bf16 sp[16 * TT];     // 16 KB bf16 unnormalized P
  __shared__ float sred[16][17];
  __shared__ float srmax[16], srcp[16];

  int mt = blockIdx.x;
  int b  = blockIdx.y;
  int g  = blockIdx.z;
  int c  = g >> 3, h = g & 7;
  size_t gbase = ((size_t)g * MROWS + (size_t)b * TT) * UPAD;
  const __bf16* qrow = qw + gbase + (size_t)mt * 16 * UPAD;
  const __bf16* krow = kb + gbase;
  const __bf16* vgb  = vt + (size_t)g * UPAD * MROWS + (size_t)b * TT;
  int w = threadIdx.x >> 5;

  // Q tile A-fragments (K = 128 -> 4 chunks), reused across all N-tiles
  v16bf afr[4];
#pragma unroll
  for (int kc = 0; kc < 4; ++kc) afr[kc] = load_a_frag(qrow + kc * 32, UPAD);

  // logits = qw @ k^T : B-frag reads rows of k (contiguous along u)
#pragma unroll
  for (int nt4 = 0; nt4 < 4; ++nt4) {
    int nt  = w + nt4 * 8;
    v8f acc = {};
#pragma unroll
    for (int kc = 0; kc < 4; ++kc) {
      v16bf bf = load_bk_frag(krow + (size_t)nt * 16 * UPAD + kc * 32, UPAD);
      acc = wmma_bf16(afr[kc], bf, acc);
    }
    store_d_f32(slog + nt * 16, TT, acc);
  }
  __syncthreads();

  // row softmax stats: thread t handles row (t&15), 32-col segment (t>>4)
  int row = threadIdx.x & 15, seg = threadIdx.x >> 4;
  float mx = -3.4e38f;
  for (int j = 0; j < 32; ++j) mx = fmaxf(mx, slog[row * TT + seg * 32 + j]);
  sred[row][seg] = mx;
  __syncthreads();
  if (threadIdx.x < 16) {
    float m2 = sred[threadIdx.x][0];
    for (int s = 1; s < 16; ++s) m2 = fmaxf(m2, sred[threadIdx.x][s]);
    srmax[threadIdx.x] = m2;
  }
  __syncthreads();
  float rm  = srmax[row];
  float sum = 0.f;
  for (int j = 0; j < 32; ++j) {
    float e = __expf(slog[row * TT + seg * 32 + j] - rm);
    sp[row * TT + seg * 32 + j] = (__bf16)e;
    sum += e;
  }
  sred[row][seg] = sum;
  __syncthreads();
  if (threadIdx.x < 16) {
    float s2 = 0.f;
    for (int s = 0; s < 16; ++s) s2 += sred[threadIdx.x][s];
    srcp[threadIdx.x] = 1.0f / s2;
  }
  __syncthreads();

  // O = P @ v : A = P (LDS), B = vT (contiguous along s per u-row)
  v8f acc = {};
#pragma unroll
  for (int kc = 0; kc < TT / 32; ++kc) {
    v16bf a  = load_a_frag(sp + kc * 32, TT);
    v16bf bf = load_bk_frag(vgb + (size_t)(w * 16) * MROWS + kc * 32, MROWS);
    acc = wmma_bf16(a, bf, acc);
  }

  // scale by 1/rowsum, write head-concatenated o (C,B,T,800), skip padded u
  int lane = threadIdx.x & 31, n = lane & 15, mh = (lane >> 4) * 8;
  int u = w * 16 + n;
  if (u < UU) {
#pragma unroll
    for (int i = 0; i < 8; ++i) {
      int m = mh + i, t = mt * 16 + m;
      float val = acc[i] * srcp[m];
      ob[(((size_t)c * BB + b) * TT + t) * OUTF + h * UU + u] = (__bf16)val;
    }
  }
}

// ---------------- output projection: z = o(8192x800) @ WO_c + bias, leaky ----------------
// WOt stored transposed [c][u:OPAD][o:OUTF]. grid: (mt 512, c 6); block 224 (7 waves)

__global__ __launch_bounds__(224) void out_kernel(
    const __bf16* __restrict__ ob, const __bf16* __restrict__ wot,
    const float* __restrict__ bias, float* __restrict__ z) {
  __shared__ alignas(16) __bf16 so[16 * OUTF];  // 25.6 KB o-tile
  int mt = blockIdx.x;
  int c  = blockIdx.y;

  stage_tile(ob + ((size_t)c * MROWS + (size_t)mt * 16) * OUTF, so,
             (16 * OUTF * 2) / 16, 224);
  __syncthreads();

  int w = threadIdx.x >> 5;
  const __bf16* Wc = wot + ((size_t)c * OPAD + w * 16) * OUTF;
  v8f acc = {};
#pragma unroll
  for (int kc = 0; kc < OUTF / 32; ++kc) {  // 25 chunks
    v16bf a = load_a_frag(so + kc * 32, OUTF);
    v16bf b = load_bk_frag(Wc + kc * 32, OUTF);
    acc = wmma_bf16(a, b, acc);
  }

  int lane = threadIdx.x & 31, n = lane & 15, mh = (lane >> 4) * 8;
  int u = w * 16 + n;
  if (u < UU) {
#pragma unroll
    for (int i = 0; i < 8; ++i) {
      int m = mh + i;
      int M = mt * 16 + m;      // flattened (b,t)
      int t = M & (TT - 1);
      float val = acc[i] + bias[t * UU + u];
      val = (val >= 0.f) ? val : 0.1f * val;
      z[((size_t)c * MROWS + M) * UU + u] = val;
    }
  }
}

// ---------------- host launcher ----------------

extern "C" void kernel_launch(void* const* d_in, const int* in_sizes, int n_in,
                              void* d_out, int out_size, void* d_ws, size_t ws_size,
                              hipStream_t stream) {
  (void)in_sizes; (void)n_in; (void)out_size; (void)ws_size;
  const float* x    = (const float*)d_in[0];
  const float* WQ   = (const float*)d_in[1];
  const float* WK   = (const float*)d_in[2];
  const float* WV   = (const float*)d_in[3];
  const float* Wd   = (const float*)d_in[4];
  const float* WO   = (const float*)d_in[5];
  const float* bias = (const float*)d_in[6];
  float* z = (float*)d_out;

  char*  ws  = (char*)d_ws;
  size_t off = 0;
  auto take = [&](size_t bytes) -> char* {
    char* p = ws + off;
    off = (off + bytes + 255) & ~(size_t)255;
    return p;
  };
  __bf16* xb  = (__bf16*)take((size_t)BB * TT * DD * 2);
  __bf16* wqd = (__bf16*)take((size_t)CC * HH * UPAD * DD * 2);   // transposed [g][u][d]
  __bf16* wkb = (__bf16*)take((size_t)CC * HH * UPAD * DD * 2);   // transposed
  __bf16* wvb = (__bf16*)take((size_t)CC * HH * UPAD * DD * 2);   // transposed
  __bf16* wot = (__bf16*)take((size_t)CC * OPAD * OUTF * 2);      // transposed
  __bf16* qw  = (__bf16*)take((size_t)CC * HH * MROWS * UPAD * 2);
  __bf16* kb  = (__bf16*)take((size_t)CC * HH * MROWS * UPAD * 2);
  __bf16* vt  = (__bf16*)take((size_t)CC * HH * UPAD * MROWS * 2);  // transposed [g][u][M]
  __bf16* ob  = (__bf16*)take((size_t)CC * MROWS * OUTF * 2);

  int nx = BB * TT * DD;
  cvt_bf16_kernel<<<(nx + 255) / 256, 256, 0, stream>>>(x, xb, nx);

  int nw = CC * HH * UPAD * DD;
  prep_wqd_kernel<<<(nw + 255) / 256, 256, 0, stream>>>(WQ, Wd, wqd);
  prep_wpad_kernel<<<(nw + 255) / 256, 256, 0, stream>>>(WK, wkb);
  prep_wpad_kernel<<<(nw + 255) / 256, 256, 0, stream>>>(WV, wvb);

  int nwo = CC * OPAD * OUTF;
  prep_wo_kernel<<<(nwo + 255) / 256, 256, 0, stream>>>(WO, wot);

  qk_kernel<<<dim3(MROWS / 16, CC * HH, 2), 256, 0, stream>>>(xb, wqd, wkb, qw, kb);
  vproj_kernel<<<dim3(MROWS / 16, CC * HH), 256, 0, stream>>>(xb, wvb, vt);
  attn_kernel<<<dim3(TT / 16, BB, CC * HH), 256, 0, stream>>>(qw, kb, vt, ob);
  out_kernel<<<dim3(MROWS / 16, CC), 224, 0, stream>>>(ob, wot, bias, z);
}
